// Encoder_67190468378802
// MI455X (gfx1250) — matrix operand order repaired and checked by reading, not compiled, not run on hardware
//
#include <hip/hip_runtime.h>
#include <hip/hip_bf16.h>

// ---------------------------------------------------------------------------
// Graph-VAE encoder for gfx1250 (MI455X):
//   conv1(GCN,128->128)+relu -> conv2(GCN,128->128)+relu -> LSTM(T=20000,H=128)
//   -> z_mean = GCN(128->64), z_log_std = GCN(128->64)
// Dense GEMMs use v_wmma_f32_16x16x32_f16 (f32 accumulate). Scatter uses native
// f32 global atomics. LSTM recurrence: one 512-thread block, Whh rows held in
// VGPRs, h/c/gates broadcast through LDS, 4-way-split accumulation to shorten
// the per-step dependent-FMA chain (the serial-latency critical path).
// ---------------------------------------------------------------------------

typedef __attribute__((ext_vector_type(16))) _Float16 v16h;
typedef __attribute__((ext_vector_type(8)))  _Float16 v8h;
typedef __attribute__((ext_vector_type(8)))  float    v8f;

#define HID 128
#define GATES 512

__device__ __forceinline__ void atomic_add_f32(float* p, float v) {
  __hip_atomic_fetch_add(p, v, __ATOMIC_RELAXED, __HIP_MEMORY_SCOPE_AGENT);
}

// ---------------- elementwise / setup kernels ----------------

__global__ void k_fill(float* __restrict__ p, float v, int n) {
  int i = blockIdx.x * blockDim.x + threadIdx.x;
  if (i < n) p[i] = v;
}

__global__ void k_deg_edges(const int* __restrict__ ei, float* __restrict__ deg, int E) {
  int i = blockIdx.x * blockDim.x + threadIdx.x;
  if (i < E) atomic_add_f32(&deg[ei[E + i]], 1.0f);   // target = edge_index[1]
}

__global__ void k_dinv(const float* __restrict__ deg, float* __restrict__ dinv, int n) {
  int i = blockIdx.x * blockDim.x + threadIdx.x;
  if (i < n) {
    float d = deg[i];
    dinv[i] = (d > 0.0f) ? rsqrtf(d) : 0.0f;
  }
}

__global__ void k_f32_to_f16(const float* __restrict__ in, _Float16* __restrict__ out, int n) {
  int i = blockIdx.x * blockDim.x + threadIdx.x;
  if (i < n) out[i] = (_Float16)in[i];
}

__global__ void k_relu(float* __restrict__ p, int n) {
  int i = blockIdx.x * blockDim.x + threadIdx.x;
  if (i < n) p[i] = fmaxf(p[i], 0.0f);
}

// Pack weight W (K x M, or M x K if transposeSrc) into the per-lane contiguous
// B-fragment layout for v_wmma_f32_16x16x32_f16:
//   Bp[(((jt*(K/32)+kt)*32 + lane)*16 + e] = W[k][m],
//   k = kt*32 + (lane>=16 ? 16:0) + e,  m = jt*16 + (lane&15)
__global__ void k_pack_b(const float* __restrict__ W, _Float16* __restrict__ Bp,
                         int K, int M, int transposeSrc) {
  int idx = blockIdx.x * blockDim.x + threadIdx.x;
  int total = K * M;
  if (idx >= total) return;
  int e    = idx & 15;
  int lane = (idx >> 4) & 31;
  int rest = idx >> 9;
  int ktiles = K >> 5;
  int kt = rest % ktiles;
  int jt = rest / ktiles;
  int n  = lane & 15;
  int hi = lane >> 4;
  int k  = kt * 32 + (hi ? 16 : 0) + e;
  int m  = jt * 16 + n;
  float v = transposeSrc ? W[(size_t)m * K + k] : W[(size_t)k * M + m];
  Bp[idx] = (_Float16)v;
}

// ---------------- WMMA GEMM: C[N,M] = A[N,K] * B[K,M], f16 in / f32 out -----
// One wave per 16x16 output tile. N,M multiples of 16; K multiple of 32.
__global__ void k_wmma_gemm(const _Float16* __restrict__ A,
                            const _Float16* __restrict__ Bp,
                            float* __restrict__ C, int K, int M) {
  const int it   = blockIdx.x;       // row tile
  const int jt   = blockIdx.y;       // col tile
  const int lane = threadIdx.x;      // 0..31
  const int mrow = lane & 15;
  const int hi   = lane >> 4;
  const int ktiles = K >> 5;

  const _Float16* Arow = A + (size_t)(it * 16 + mrow) * K;
  v8f acc = {};

  for (int kt = 0; kt < ktiles; ++kt) {
    const int kbase = kt * 32;
    // A fragment: 16x32 f16, contiguous k = 16*hi + e (matches packed B)
    v8h alo = *reinterpret_cast<const v8h*>(Arow + kbase + (hi ? 16 : 0));
    v8h ahi = *reinterpret_cast<const v8h*>(Arow + kbase + (hi ? 24 : 8));
    v16h a;
#pragma unroll
    for (int e = 0; e < 8; ++e) { a[e] = alo[e]; a[e + 8] = ahi[e]; }

    // B fragment: pre-packed contiguous per lane -> two 16B loads
    const _Float16* bptr = Bp + (((size_t)jt * ktiles + kt) * 32 + lane) * 16;
    v8h blo = *reinterpret_cast<const v8h*>(bptr);
    v8h bhi = *reinterpret_cast<const v8h*>(bptr + 8);
    v16h b;
#pragma unroll
    for (int e = 0; e < 8; ++e) { b[e] = blo[e]; b[e + 8] = bhi[e]; }

    acc = __builtin_amdgcn_wmma_f32_16x16x32_f16(
        /*neg_a=*/false, a, /*neg_b=*/false, b,
        /*c_mod=*/(short)0, acc, /*reuse_a=*/false, /*reuse_b=*/false);
  }

  // C/D layout: lane holds column n=lane&15; VGPR r holds row m = r + hi*8
#pragma unroll
  for (int r = 0; r < 8; ++r) {
    int row = it * 16 + r + hi * 8;
    C[(size_t)row * M + jt * 16 + mrow] = acc[r];
  }
}

// ---------------- GCN scatter phase ----------------

// out[i,m] = bias[m] + G[i,m] * dinv[i]^2   (self-loop term + bias)
__global__ void k_conv_init(const float* __restrict__ G, const float* __restrict__ dinv,
                            const float* __restrict__ bias, float* __restrict__ out,
                            int N, int M) {
  int idx = blockIdx.x * blockDim.x + threadIdx.x;
  if (idx >= N * M) return;
  int i = idx / M, m = idx - i * M;
  float di = dinv[i];
  out[idx] = bias[m] + G[idx] * di * di;
}

// one wave per edge: out[col] += G[row] * dinv[row]*dinv[col]
__global__ void k_conv_edges(const float* __restrict__ G, const float* __restrict__ dinv,
                             const int* __restrict__ ei, float* __restrict__ out,
                             int E, int M) {
  int gid  = blockIdx.x * blockDim.x + threadIdx.x;
  int e    = gid >> 5;
  int lane = gid & 31;
  if (e >= E) return;
  int r = ei[e];            // source
  int c = ei[E + e];        // target
  float norm = dinv[r] * dinv[c];
  const float* src = G + (size_t)r * M;
  float*       dst = out + (size_t)c * M;
  for (int j = lane; j < M; j += 32)
    atomic_add_f32(dst + j, src[j] * norm);
}

// ---------------- LSTM recurrence ----------------
// xg: [T, 512] precomputed x@Wih^T (bias added here). One block, 512 threads.
// Thread m owns Whh row m in VGPRs; h/c/gates live in LDS. The h.w dot product
// uses 4 interleaved accumulators (dependent-chain depth 32 instead of 128)
// and float4 LDS reads (ds_load_b128) to minimize per-step serial latency.
__global__ void __launch_bounds__(GATES)
k_lstm(const float* __restrict__ xg, const float* __restrict__ Whh,
       const float* __restrict__ bih, const float* __restrict__ bhh,
       float* __restrict__ hs, int T) {
  __shared__ __align__(16) float h_s[HID];
  __shared__ __align__(16) float c_s[HID];
  __shared__ __align__(16) float g_s[GATES];

  const int m = threadIdx.x;          // 0..511
  float w[HID];
#pragma unroll
  for (int k = 0; k < HID; ++k) w[k] = Whh[(size_t)m * HID + k];
  const float bias = bih[m] + bhh[m];
  const int gate = m >> 7;            // 0=i 1=f 2=g 3=o (torch order)

  if (m < HID) { h_s[m] = 0.0f; c_s[m] = 0.0f; }
  __syncthreads();

  for (int t = 0; t < T; ++t) {
    const float xt = xg[(size_t)t * GATES + m];
    float a0 = 0.0f, a1 = 0.0f, a2 = 0.0f, a3 = 0.0f;
#pragma unroll
    for (int k = 0; k < HID; k += 4) {
      float4 h4 = *reinterpret_cast<const float4*>(&h_s[k]);
      a0 = fmaf(h4.x, w[k + 0], a0);
      a1 = fmaf(h4.y, w[k + 1], a1);
      a2 = fmaf(h4.z, w[k + 2], a2);
      a3 = fmaf(h4.w, w[k + 3], a3);
    }
    const float acc = xt + bias + ((a0 + a1) + (a2 + a3));
    g_s[m] = (gate == 2) ? tanhf(acc) : 1.0f / (1.0f + __expf(-acc));
    __syncthreads();
    if (m < HID) {
      float i  = g_s[m];
      float f  = g_s[HID + m];
      float gg = g_s[2 * HID + m];
      float o  = g_s[3 * HID + m];
      float c  = f * c_s[m] + i * gg;
      c_s[m] = c;
      float h = o * tanhf(c);
      h_s[m] = h;
      hs[(size_t)t * HID + m] = h;
    }
    __syncthreads();
  }
}

// ---------------------------------------------------------------------------

static inline int cdiv(int a, int b) { return (a + b - 1) / b; }

extern "C" void kernel_launch(void* const* d_in, const int* in_sizes, int n_in,
                              void* d_out, int out_size, void* d_ws, size_t ws_size,
                              hipStream_t stream) {
  const float* x    = (const float*)d_in[0];
  const int*   ei   = (const int*)d_in[1];
  const float* W1   = (const float*)d_in[2];
  const float* b1   = (const float*)d_in[3];
  const float* W2   = (const float*)d_in[4];
  const float* b2   = (const float*)d_in[5];
  const float* Wih  = (const float*)d_in[6];
  const float* Whh  = (const float*)d_in[7];
  const float* bih  = (const float*)d_in[8];
  const float* bhh  = (const float*)d_in[9];
  const float* Wm   = (const float*)d_in[10];
  const float* bm   = (const float*)d_in[11];
  const float* Wl   = (const float*)d_in[12];
  const float* bl   = (const float*)d_in[13];

  const int N   = in_sizes[0] / HID;      // 20000 (multiple of 16)
  const int E   = in_sizes[1] / 2;        // 640000
  const int LAT = in_sizes[11];           // 64

  // ---- workspace layout (512B-aligned offsets) ----
  char* ws = (char*)d_ws;
  float*    dinv = (float*)(ws + 0);                     // N floats
  _Float16* A16  = (_Float16*)(ws + 81920);              // N*128 halves
  _Float16* B16  = (_Float16*)(ws + 5201920);            // up to 128*512 halves
  float*    G    = (float*)(ws + 5332992);               // up to N*512 floats
  float*    H1   = (float*)(ws + 46292992);              // N*128 floats
  float*    H2   = (float*)(ws + 56532992);              // N*128 floats
  float*    deg  = G;                                    // temp reuse before GEMMs

  float* zmean = (float*)d_out;
  float* zlog  = zmean + (size_t)N * LAT;

  const int TB = 256;

  // ---- degree / normalization (self-loop => deg starts at 1) ----
  k_fill<<<cdiv(N, TB), TB, 0, stream>>>(deg, 1.0f, N);
  k_deg_edges<<<cdiv(E, TB), TB, 0, stream>>>(ei, deg, E);
  k_dinv<<<cdiv(N, TB), TB, 0, stream>>>(deg, dinv, N);

  // ---- conv1: H1 = relu(GCN(x, W1, b1)) ----
  k_f32_to_f16<<<cdiv(N * HID, TB), TB, 0, stream>>>(x, A16, N * HID);
  k_pack_b<<<cdiv(HID * HID, TB), TB, 0, stream>>>(W1, B16, HID, HID, 0);
  {
    dim3 grid(N / 16, HID / 16);
    k_wmma_gemm<<<grid, 32, 0, stream>>>(A16, B16, G, HID, HID);
  }
  k_conv_init<<<cdiv(N * HID, TB), TB, 0, stream>>>(G, dinv, b1, H1, N, HID);
  k_conv_edges<<<cdiv(E * 32, TB), TB, 0, stream>>>(G, dinv, ei, H1, E, HID);
  k_relu<<<cdiv(N * HID, TB), TB, 0, stream>>>(H1, N * HID);

  // ---- conv2: H2 = relu(GCN(H1, W2, b2)) ----
  k_f32_to_f16<<<cdiv(N * HID, TB), TB, 0, stream>>>(H1, A16, N * HID);
  k_pack_b<<<cdiv(HID * HID, TB), TB, 0, stream>>>(W2, B16, HID, HID, 0);
  {
    dim3 grid(N / 16, HID / 16);
    k_wmma_gemm<<<grid, 32, 0, stream>>>(A16, B16, G, HID, HID);
  }
  k_conv_init<<<cdiv(N * HID, TB), TB, 0, stream>>>(G, dinv, b2, H2, N, HID);
  k_conv_edges<<<cdiv(E * 32, TB), TB, 0, stream>>>(G, dinv, ei, H2, E, HID);
  k_relu<<<cdiv(N * HID, TB), TB, 0, stream>>>(H2, N * HID);

  // ---- LSTM input projection: G = H2 @ Wih^T  [N, 512] ----
  k_f32_to_f16<<<cdiv(N * HID, TB), TB, 0, stream>>>(H2, A16, N * HID);
  k_pack_b<<<cdiv(HID * GATES, TB), TB, 0, stream>>>(Wih, B16, HID, GATES, 1);
  {
    dim3 grid(N / 16, GATES / 16);
    k_wmma_gemm<<<grid, 32, 0, stream>>>(A16, B16, G, HID, GATES);
  }

  // ---- LSTM recurrence: hs -> H1 ----
  k_lstm<<<1, GATES, 0, stream>>>(G, Whh, bih, bhh, H1, N);

  // ---- z_mean = GCN(hs, Wm, bm) ----
  k_f32_to_f16<<<cdiv(N * HID, TB), TB, 0, stream>>>(H1, A16, N * HID);
  k_pack_b<<<cdiv(HID * LAT, TB), TB, 0, stream>>>(Wm, B16, HID, LAT, 0);
  {
    dim3 grid(N / 16, LAT / 16);
    k_wmma_gemm<<<grid, 32, 0, stream>>>(A16, B16, G, HID, LAT);
  }
  k_conv_init<<<cdiv(N * LAT, TB), TB, 0, stream>>>(G, dinv, bm, zmean, N, LAT);
  k_conv_edges<<<cdiv(E * 32, TB), TB, 0, stream>>>(G, dinv, ei, zmean, E, LAT);

  // ---- z_log_std = GCN(hs, Wl, bl) ----
  k_pack_b<<<cdiv(HID * LAT, TB), TB, 0, stream>>>(Wl, B16, HID, LAT, 0);
  {
    dim3 grid(N / 16, LAT / 16);
    k_wmma_gemm<<<grid, 32, 0, stream>>>(A16, B16, G, HID, LAT);
  }
  k_conv_init<<<cdiv(N * LAT, TB), TB, 0, stream>>>(G, dinv, bl, zlog, N, LAT);
  k_conv_edges<<<cdiv(E * 32, TB), TB, 0, stream>>>(G, dinv, ei, zlog, E, LAT);

  (void)n_in; (void)out_size; (void)ws_size;
}
